// LePEAttention_45930380263554
// MI455X (gfx1250) — matrix operand order repaired and checked by reading, not compiled, not run on hardware
//
#include <hip/hip_runtime.h>

typedef __attribute__((ext_vector_type(16))) _Float16 v16h;
typedef __attribute__((ext_vector_type(8)))  _Float16 v8h;
typedef __attribute__((ext_vector_type(8)))  float    v8f;

// Problem constants (match reference)
constexpr int Bb   = 8;
constexpr int Hh   = 64;
constexpr int Ww   = 64;
constexpr int Cc   = 256;
constexpr int HD   = 32;     // head dim
constexpr int WSP  = 8;      // window width
constexpr int L    = Hh * Ww;           // 4096 tokens per image
constexpr int S    = 64 * 8;            // 512 tokens per window
constexpr float SCALE = 0.1767766952966369f; // 32^-0.5

// One block = one (window, head) problem x 128 query rows (8 waves x 16 rows).
// 512 problems x 4 query-blocks = 2048 blocks.
__global__ __launch_bounds__(256) void cswin_lepe_attn(
    const float* __restrict__ x,      // [3, B, L, C]  (q,k,v)
    const float* __restrict__ convw,  // [C, 1, 3, 3] -> [C,9]
    const float* __restrict__ convb,  // [C]
    float* __restrict__ out)          // [B, L, C]
{
    __shared__ _Float16 Ksh[32][32];       // key chunk, row-major [key][d]
    __shared__ _Float16 VshT[32][32];      // value chunk, transposed [d][row]
    __shared__ _Float16 Psh[8][16][32];    // per-wave P relayout buffer

    const int tid  = threadIdx.x;
    const int lane = tid & 31;
    const int wid  = tid >> 5;          // wave in block, 0..7
    const int bx   = blockIdx.x;
    const int p    = bx >> 2;           // problem id 0..511
    const int qb   = bx & 3;            // query block 0..3
    const int widx = p >> 3;            // window 0..63
    const int hd   = p & 7;             // head 0..7
    const int b    = widx >> 3;         // batch
    const int wb   = widx & 7;          // window column block
    const int qtile = qb * 8 + wid;     // 0..31 (16 queries each)

    const float* Qg = x;
    const float* Kg = x + (size_t)Bb * L * Cc;
    const float* Vg = x + (size_t)2 * Bb * L * Cc;

    const int hw16 = lane >> 4;         // 0 or 1 (half-wave)
    const int l15  = lane & 15;

    // token s in window -> flat [B,L,C] base index of this head's 32 channels
    auto rowbase = [&](int s) -> size_t {
        int h  = s >> 3;
        int wi = s & 7;
        return ((size_t)b * L + (size_t)h * Ww + (size_t)(wb * WSP + wi)) * Cc
               + (size_t)hd * HD;
    };

    // ---- Q tile (16x32) in WMMA A-layout: lane half selects K sub-blocks
    v16h qA;
    {
        const int m  = l15;
        const int s  = qtile * 16 + m;
        const size_t base = rowbase(s);
        const int kb = hw16 * 16;  // lanes 0-15: K {0..15}; 16-31: {16..31}
        #pragma unroll
        for (int e = 0; e < 8; ++e)
            qA[e] = (_Float16)(Qg[base + kb + e] * SCALE);
        #pragma unroll
        for (int e = 0; e < 8; ++e)
            qA[8 + e] = (_Float16)(Qg[base + kb + 8 + e] * SCALE);
    }

    v8f acc0 = {};  // O columns d = 0..15   (C-layout)
    v8f acc1 = {};  // O columns d = 16..31
    float mrow[8], lrow[8];
    #pragma unroll
    for (int r = 0; r < 8; ++r) { mrow[r] = -1e30f; lrow[r] = 0.0f; }

    const v8f zero = {};

    for (int kc = 0; kc < 16; ++kc) {
        __syncthreads();   // previous chunk fully consumed
        // -------- cooperative stage of 32 keys + 32 values (f32 -> f16)
        {
            const int row = tid >> 3;         // 0..31
            const int d0  = (tid & 7) * 4;    // 0,4,...,28
            const int s   = kc * 32 + row;
            const size_t base = rowbase(s);
            float k0 = Kg[base + d0 + 0], k1 = Kg[base + d0 + 1];
            float k2 = Kg[base + d0 + 2], k3 = Kg[base + d0 + 3];
            Ksh[row][d0 + 0] = (_Float16)k0;
            Ksh[row][d0 + 1] = (_Float16)k1;
            Ksh[row][d0 + 2] = (_Float16)k2;
            Ksh[row][d0 + 3] = (_Float16)k3;
            float v0 = Vg[base + d0 + 0], v1 = Vg[base + d0 + 1];
            float v2 = Vg[base + d0 + 2], v3 = Vg[base + d0 + 3];
            VshT[d0 + 0][row] = (_Float16)v0;
            VshT[d0 + 1][row] = (_Float16)v1;
            VshT[d0 + 2][row] = (_Float16)v2;
            VshT[d0 + 3][row] = (_Float16)v3;
            if (kc + 1 < 16) {   // prefetch next chunk (global_prefetch_b8)
                const size_t nb = rowbase((kc + 1) * 32 + row) + d0;
                __builtin_prefetch(&Kg[nb], 0, 3);
                __builtin_prefetch(&Vg[nb], 0, 3);
            }
        }
        __syncthreads();   // chunk staged

        // -------- B operands from LDS (ISA B-layout: lane=column, K split by half-wave)
        const int db = hw16 * 16;
        v16h kB0 = *(const v16h*)&Ksh[l15][db];        // keys 0..15 of chunk
        v16h kB1 = *(const v16h*)&Ksh[16 + l15][db];   // keys 16..31 of chunk
        v16h vB0 = *(const v16h*)&VshT[l15][db];       // V columns d=0..15
        v16h vB1 = *(const v16h*)&VshT[16 + l15][db];  // V columns d=16..31

        // -------- scores: S = Q . K^T  (two 16x16 tiles)
        v8f s0 = __builtin_amdgcn_wmma_f32_16x16x32_f16(
            false, qA, false, kB0, (short)0, zero, false, false);
        v8f s1 = __builtin_amdgcn_wmma_f32_16x16x32_f16(
            false, qA, false, kB1, (short)0, zero, false, false);

        // -------- online softmax (row stats live within 16-lane groups)
        float p0[8], p1[8];
        #pragma unroll
        for (int r = 0; r < 8; ++r) {
            float mx = fmaxf(s0[r], s1[r]);
            mx = fmaxf(mx, __shfl_xor(mx, 1));
            mx = fmaxf(mx, __shfl_xor(mx, 2));
            mx = fmaxf(mx, __shfl_xor(mx, 4));
            mx = fmaxf(mx, __shfl_xor(mx, 8));
            const float mnew = fmaxf(mrow[r], mx);
            const float corr = __expf(mrow[r] - mnew);
            mrow[r] = mnew;
            const float e0 = __expf(s0[r] - mnew);
            const float e1 = __expf(s1[r] - mnew);
            p0[r] = e0; p1[r] = e1;
            float rs = e0 + e1;
            rs += __shfl_xor(rs, 1);
            rs += __shfl_xor(rs, 2);
            rs += __shfl_xor(rs, 4);
            rs += __shfl_xor(rs, 8);
            lrow[r] = lrow[r] * corr + rs;
            acc0[r] *= corr;
            acc1[r] *= corr;
        }

        // -------- relayout P: C-layout -> A-layout via per-wave LDS region
        #pragma unroll
        for (int r = 0; r < 8; ++r) {
            const int m = r + hw16 * 8;
            Psh[wid][m][l15]      = (_Float16)p0[r];
            Psh[wid][m][16 + l15] = (_Float16)p1[r];
        }
        // per-wave LDS is in-order; wait for stores + block compiler reordering
        asm volatile("s_wait_dscnt 0" ::: "memory");
        v16h pA;
        {
            const int m  = l15;
            const int kb = hw16 * 16;
            v8h c0 = *(const v8h*)&Psh[wid][m][kb];
            v8h c1 = *(const v8h*)&Psh[wid][m][kb + 8];
            #pragma unroll
            for (int e = 0; e < 8; ++e) { pA[e] = c0[e]; pA[8 + e] = c1[e]; }
        }

        // -------- O += P . V  (two d-halves)
        acc0 = __builtin_amdgcn_wmma_f32_16x16x32_f16(
            false, pA, false, vB0, (short)0, acc0, false, false);
        acc1 = __builtin_amdgcn_wmma_f32_16x16x32_f16(
            false, pA, false, vB1, (short)0, acc1, false, false);
    }

    // -------- epilogue: normalize, add LePE (depthwise 3x3 per window), scatter
    // per-lane channels: c = hd*32 + t*16 + l15  (t = 0 -> acc0, t = 1 -> acc1)
    float wv[2][9], bb[2];
    #pragma unroll
    for (int t = 0; t < 2; ++t) {
        const int c = hd * HD + t * 16 + l15;
        bb[t] = convb[c];
        #pragma unroll
        for (int j = 0; j < 9; ++j) wv[t][j] = convw[c * 9 + j];
    }

    #pragma unroll
    for (int r = 0; r < 8; ++r) {
        const int m  = r + hw16 * 8;
        const int s  = qtile * 16 + m;
        const int sh = s >> 3;
        const int sw = s & 7;
        const float inv = 1.0f / lrow[r];
        #pragma unroll
        for (int t = 0; t < 2; ++t) {
            const int c = hd * HD + t * 16 + l15;
            float o = (t == 0 ? acc0[r] : acc1[r]) * inv;
            float lep = bb[t];
            #pragma unroll
            for (int dh = -1; dh <= 1; ++dh) {
                const int hh = sh + dh;
                if ((unsigned)hh < (unsigned)Hh) {
                    #pragma unroll
                    for (int dw = -1; dw <= 1; ++dw) {
                        const int ww = sw + dw;
                        if ((unsigned)ww < (unsigned)WSP) {
                            const size_t va =
                                ((size_t)b * L + (size_t)hh * Ww +
                                 (size_t)(wb * WSP + ww)) * Cc + c;
                            lep += Vg[va] * wv[t][(dh + 1) * 3 + (dw + 1)];
                        }
                    }
                }
            }
            const size_t oa = ((size_t)b * L + (size_t)sh * Ww +
                               (size_t)(wb * WSP + sw)) * Cc + c;
            out[oa] = o + lep;
        }
    }
}

extern "C" void kernel_launch(void* const* d_in, const int* in_sizes, int n_in,
                              void* d_out, int out_size, void* d_ws, size_t ws_size,
                              hipStream_t stream) {
    (void)in_sizes; (void)n_in; (void)out_size; (void)d_ws; (void)ws_size;
    const float* x     = (const float*)d_in[0];
    const float* convw = (const float*)d_in[1];
    const float* convb = (const float*)d_in[2];
    float* out = (float*)d_out;

    dim3 grid(2048);   // 512 problems x 4 query blocks
    dim3 block(256);   // 8 waves (wave32)
    hipLaunchKernelGGL(cswin_lepe_attn, grid, block, 0, stream,
                       x, convw, convb, out);
}
